// MyLinear_48679159333251
// MI455X (gfx1250) — compile-verified, matrix operand order checked by reading, not run on hardware
//
#include <hip/hip_runtime.h>
#include <hip/hip_bf16.h>
#include <math.h>

// Problem constants (match the reference)
#define BDIM   8192
#define INDIM  4096
#define OUTDIM 4096
#define EARLY_K 32
#define STAT_BOUND (-1.6448536269514722f)

typedef __attribute__((ext_vector_type(16))) __bf16 v16bf;
typedef __attribute__((ext_vector_type(8)))  __bf16 v8bf;
typedef __attribute__((ext_vector_type(4)))  __bf16 v4bf;
typedef __attribute__((ext_vector_type(8)))  float  v8f;
typedef __attribute__((ext_vector_type(4)))  float  v4f;
typedef __attribute__((ext_vector_type(2)))  float  v2f;
typedef __attribute__((ext_vector_type(4)))  int    v4i;

#define TILE_M 128
#define TILE_N 128
#define KT      32          // K per main-loop step (one bf16 WMMA K-depth)
#define LDS_STRIDE 40       // 32 + 8 pad bf16 elems: 16B-aligned rows, conflict-free banks
#define PLANE_ELEMS (TILE_M * LDS_STRIDE)

// ---------------------------------------------------------------------------
// CDNA5 async global->LDS path (ASYNCcnt-tracked, no VGPR round trip).
// Builtin signature (probe-confirmed): (v4i AS1*, v4i AS3*, Ii offset, Ii cpol)
// ---------------------------------------------------------------------------
#if defined(__has_builtin)
#if __has_builtin(__builtin_amdgcn_global_load_async_to_lds_b128)
#define HAVE_ASYNC_LDS 1
#endif
#if __has_builtin(__builtin_amdgcn_s_wait_asynccnt)
#define HAVE_WAIT_ASYNC 1
#endif
#endif

#ifdef HAVE_ASYNC_LDS
typedef __attribute__((address_space(1))) v4i* gv4i_p;   // global int4*
typedef __attribute__((address_space(3))) v4i* lv4i_p;   // LDS int4*
__device__ __forceinline__ void stage16(const void* g, void* l) {
  __builtin_amdgcn_global_load_async_to_lds_b128(
      (gv4i_p)(unsigned long long)(uintptr_t)g,
      (lv4i_p)(unsigned int)(uintptr_t)l,
      0, 0);
}
__device__ __forceinline__ void stageWait() {
#ifdef HAVE_WAIT_ASYNC
  __builtin_amdgcn_s_wait_asynccnt(0);
#else
  asm volatile("s_wait_asynccnt 0x0" ::: "memory");
#endif
}
#else
__device__ __forceinline__ void stage16(const void* g, void* l) {
  *(v8bf*)l = *(const v8bf*)g;   // global_load_b128 + ds_store_b128 fallback
}
__device__ __forceinline__ void stageWait() {}
#endif

// Load one 16x32 bf16 A/B fragment from LDS per the CDNA5 16-bit layout:
// lane L (0-15):  row = base+L, K = kp..kp+7 in v0..v3, K = kp+16..kp+23 in v4..v7
// lane L+16:      same row, kp offset by +8
__device__ __forceinline__ v16bf ldsFrag(const __bf16* s, int row, int kp) {
  const __bf16* p = s + row * LDS_STRIDE + kp;
  v8bf a = *(const v8bf*)(p);
  v8bf b = *(const v8bf*)(p + 16);
  return __builtin_shufflevector(a, b, 0,1,2,3,4,5,6,7,8,9,10,11,12,13,14,15);
}

// ---------------------------------------------------------------------------
// Kernel 0: split f32 -> bf16 hi/lo planes (one-time, bandwidth-bound pass)
// ---------------------------------------------------------------------------
__global__ __launch_bounds__(256) void precvt(const float* __restrict__ src,
                                              __bf16* __restrict__ hi,
                                              __bf16* __restrict__ lo, int n4) {
  int i = blockIdx.x * blockDim.x + threadIdx.x;
  int stride = gridDim.x * blockDim.x;
  for (; i < n4; i += stride) {
    v4f v = *(const v4f*)(src + 4 * (size_t)i);
    v4bf h, l;
#pragma unroll
    for (int j = 0; j < 4; ++j) {
      __bf16 hh = (__bf16)v[j];
      h[j] = hh;
      l[j] = (__bf16)(v[j] - (float)hh);
    }
    *(v4bf*)(hi + 4 * (size_t)i) = h;
    *(v4bf*)(lo + 4 * (size_t)i) = l;
  }
}

// ---------------------------------------------------------------------------
// Shared device helpers for the early (exact f32) window + mask + epilogue
// ---------------------------------------------------------------------------
__device__ __forceinline__ void earlyWindow(const float* __restrict__ x,
                                            const float* __restrict__ W,
                                            int tM, int tN, int wM, int wN,
                                            int laneLow, int laneHi,
                                            v8f acc[2][4], v8f cs[2][4]) {
#pragma unroll
  for (int kk = 0; kk < EARLY_K; kk += 4) {
    v2f a[2], b[4];
#pragma unroll
    for (int mt = 0; mt < 2; ++mt) {
      int m = tM + wM + mt * 16 + laneLow;
      a[mt] = *(const v2f*)(x + (size_t)m * INDIM + kk + 2 * laneHi);
    }
#pragma unroll
    for (int nt = 0; nt < 4; ++nt) {
      int n = tN + wN + nt * 16 + laneLow;
      b[nt] = *(const v2f*)(W + (size_t)n * INDIM + kk + 2 * laneHi);
    }
#pragma unroll
    for (int mt = 0; mt < 2; ++mt) {
      v2f a2 = a[mt] * a[mt];
#pragma unroll
      for (int nt = 0; nt < 4; ++nt) {
        acc[mt][nt] = __builtin_amdgcn_wmma_f32_16x16x4_f32(
            false, a[mt], false, b[nt], (short)0, acc[mt][nt], false, false);
        cs[mt][nt] = __builtin_amdgcn_wmma_f32_16x16x4_f32(
            false, a2, false, b[nt] * b[nt], (short)0, cs[mt][nt], false, false);
      }
    }
  }
}

__device__ __forceinline__ unsigned long long makeMask(const v8f acc[2][4],
                                                       const v8f cs[2][4]) {
  unsigned long long passed = 0ull;
#pragma unroll
  for (int mt = 0; mt < 2; ++mt)
#pragma unroll
    for (int nt = 0; nt < 4; ++nt)
#pragma unroll
      for (int e = 0; e < 8; ++e) {
        float y1   = acc[mt][nt][e];
        float mean = y1 * (1.0f / EARLY_K);
        float var  = cs[mt][nt][e] * (1.0f / EARLY_K) - mean * mean;
        float stat = mean / sqrtf(var * (1.0f / EARLY_K));
        if (stat < STAT_BOUND)
          passed |= 1ull << ((mt * 4 + nt) * 8 + e);
      }
  return passed;
}

__device__ __forceinline__ void storeOut(float* __restrict__ out,
                                         const v8f acc[2][4],
                                         unsigned long long passed,
                                         int tM, int tN, int wM, int wN,
                                         int laneLow, int laneHi) {
#pragma unroll
  for (int mt = 0; mt < 2; ++mt)
#pragma unroll
    for (int nt = 0; nt < 4; ++nt) {
      int n = tN + wN + nt * 16 + laneLow;
#pragma unroll
      for (int e = 0; e < 8; ++e) {
        int m = tM + wM + mt * 16 + laneHi * 8 + e;
        float v = ((passed >> ((mt * 4 + nt) * 8 + e)) & 1ull) ? 0.0f
                                                               : acc[mt][nt][e];
        out[(size_t)m * OUTDIM + n] = v;
      }
    }
}

// ---------------------------------------------------------------------------
// Kernel 1 (fast path): GEMM from pre-converted bf16 hi/lo planes.
// Double-buffered: async DMA for tile i+1 overlaps the 24 WMMAs of tile i.
// ---------------------------------------------------------------------------
__global__ __launch_bounds__(256, 1) void gemm_planes(
    const __bf16* __restrict__ xhi, const __bf16* __restrict__ xlo,
    const __bf16* __restrict__ whi, const __bf16* __restrict__ wlo,
    const float* __restrict__ x, const float* __restrict__ W,
    float* __restrict__ out) {
  // [buffer][plane: Ahi, Alo, Bhi, Blo][128 rows x 40 bf16]  = 80 KB total
  __shared__ __bf16 smem[2][4][PLANE_ELEMS];

  const int t       = threadIdx.x;
  const int lane    = t & 31;
  const int wave    = t >> 5;
  const int laneLow = lane & 15;
  const int laneHi  = lane >> 4;
  const int kp      = laneHi * 8;

  const int tM = blockIdx.x * TILE_M;
  const int tN = blockIdx.y * TILE_N;
  const int wM = (wave & 3) * 32;
  const int wN = (wave >> 2) * 64;

  const __bf16* gp[4];
  gp[0] = xhi + (size_t)tM * INDIM;
  gp[1] = xlo + (size_t)tM * INDIM;
  gp[2] = whi + (size_t)tN * INDIM;
  gp[3] = wlo + (size_t)tN * INDIM;

  // Stage one 128x32 tile of each plane into smem[buf] for K-offset k0.
  // 512 b128 chunks per plane (128 rows x 4 chunks); 2 per thread per plane.
  auto stageTile = [&](int buf, int k0) {
#pragma unroll
    for (int p = 0; p < 4; ++p)
#pragma unroll
      for (int i = 0; i < 2; ++i) {
        int c   = t + 256 * i;       // 0..511
        int row = c >> 2;
        int kc  = (c & 3) << 3;      // bf16 elems
        stage16(gp[p] + (size_t)row * INDIM + k0 + kc,
                &smem[buf][p][row * LDS_STRIDE + kc]);
      }
  };

  // Kick off DMA of the first main-loop tile; it overlaps the early window.
  stageTile(0, EARLY_K);

  v8f acc[2][4], cs[2][4];
  {
    v8f z = {};
#pragma unroll
    for (int mt = 0; mt < 2; ++mt)
#pragma unroll
      for (int nt = 0; nt < 4; ++nt) { acc[mt][nt] = z; cs[mt][nt] = z; }
  }

  // Phase 1: exact-f32 K=0..31 via V_WMMA_F32_16X16X4_F32 (mask must be stable)
  earlyWindow(x, W, tM, tN, wM, wN, laneLow, laneHi, acc, cs);
  const unsigned long long passed = makeMask(acc, cs);

  // Phase 2: K = 32..4095, bf16x3 split-precision WMMA, software-pipelined.
  const int nIter = (INDIM - EARLY_K) / KT;   // 127
  for (int i = 0; i < nIter; ++i) {
    const int p = i & 1;
    stageWait();       // own async writes of smem[p] (issued last iter) done
    __syncthreads();   // publish smem[p]; all waves done reading smem[1-p]
    if (i + 1 < nIter)
      stageTile(1 - p, EARLY_K + (i + 1) * KT);  // DMA overlaps compute below

    const __bf16* sAhi = smem[p][0];
    const __bf16* sAlo = smem[p][1];
    const __bf16* sBhi = smem[p][2];
    const __bf16* sBlo = smem[p][3];

    v16bf ah[2], al[2], bh[4], bl[4];
#pragma unroll
    for (int mt = 0; mt < 2; ++mt) {
      int r = wM + mt * 16 + laneLow;
      ah[mt] = ldsFrag(sAhi, r, kp);
      al[mt] = ldsFrag(sAlo, r, kp);
    }
#pragma unroll
    for (int nt = 0; nt < 4; ++nt) {
      int r = wN + nt * 16 + laneLow;
      bh[nt] = ldsFrag(sBhi, r, kp);
      bl[nt] = ldsFrag(sBlo, r, kp);
    }
#pragma unroll
    for (int mt = 0; mt < 2; ++mt)
#pragma unroll
      for (int nt = 0; nt < 4; ++nt) {
        acc[mt][nt] = __builtin_amdgcn_wmma_f32_16x16x32_bf16(
            false, ah[mt], false, bh[nt], (short)0, acc[mt][nt], false, false);
        acc[mt][nt] = __builtin_amdgcn_wmma_f32_16x16x32_bf16(
            false, ah[mt], false, bl[nt], (short)0, acc[mt][nt], false, false);
        acc[mt][nt] = __builtin_amdgcn_wmma_f32_16x16x32_bf16(
            false, al[mt], false, bh[nt], (short)0, acc[mt][nt], false, false);
      }
  }

  storeOut(out, acc, passed, tM, tN, wM, wN, laneLow, laneHi);
}

// ---------------------------------------------------------------------------
// Kernel 2 (fallback if workspace too small): fused on-the-fly conversion
// ---------------------------------------------------------------------------
__global__ __launch_bounds__(256, 1) void fused_seq_linear(
    const float* __restrict__ x, const float* __restrict__ W,
    float* __restrict__ out) {
  __shared__ __bf16 sAhi[PLANE_ELEMS];
  __shared__ __bf16 sAlo[PLANE_ELEMS];
  __shared__ __bf16 sBhi[PLANE_ELEMS];
  __shared__ __bf16 sBlo[PLANE_ELEMS];

  const int t       = threadIdx.x;
  const int lane    = t & 31;
  const int wave    = t >> 5;
  const int laneLow = lane & 15;
  const int laneHi  = lane >> 4;
  const int kp      = laneHi * 8;

  const int tM = blockIdx.x * TILE_M;
  const int tN = blockIdx.y * TILE_N;
  const int wM = (wave & 3) * 32;
  const int wN = (wave >> 2) * 64;

  v8f acc[2][4], cs[2][4];
  {
    v8f z = {};
#pragma unroll
    for (int mt = 0; mt < 2; ++mt)
#pragma unroll
      for (int nt = 0; nt < 4; ++nt) { acc[mt][nt] = z; cs[mt][nt] = z; }
  }

  earlyWindow(x, W, tM, tN, wM, wN, laneLow, laneHi, acc, cs);
  const unsigned long long passed = makeMask(acc, cs);

  const float* xTile = x + (size_t)tM * INDIM;
  const float* wTile = W + (size_t)tN * INDIM;

  for (int k0 = EARLY_K; k0 < INDIM; k0 += KT) {
    __syncthreads();
#pragma unroll
    for (int i = 0; i < 4; ++i) {
      int c   = t + 256 * i;
      int row = c >> 3;
      int kc  = (c & 7) << 2;
      const float* xp = xTile + (size_t)row * INDIM + k0 + kc;
      const float* wp = wTile + (size_t)row * INDIM + k0 + kc;
      v4f xa = *(const v4f*)xp;
      v4f wa = *(const v4f*)wp;
      if (k0 + KT < INDIM) {
        __builtin_prefetch(xp + KT, 0, 0);
        __builtin_prefetch(wp + KT, 0, 0);
      }
      v4bf xh, xl, wh, wl;
#pragma unroll
      for (int j = 0; j < 4; ++j) {
        __bf16 h = (__bf16)xa[j]; xh[j] = h; xl[j] = (__bf16)(xa[j] - (float)h);
        __bf16 g = (__bf16)wa[j]; wh[j] = g; wl[j] = (__bf16)(wa[j] - (float)g);
      }
      *(v4bf*)(sAhi + row * LDS_STRIDE + kc) = xh;
      *(v4bf*)(sAlo + row * LDS_STRIDE + kc) = xl;
      *(v4bf*)(sBhi + row * LDS_STRIDE + kc) = wh;
      *(v4bf*)(sBlo + row * LDS_STRIDE + kc) = wl;
    }
    __syncthreads();

    v16bf ah[2], al[2], bh[4], bl[4];
#pragma unroll
    for (int mt = 0; mt < 2; ++mt) {
      int r = wM + mt * 16 + laneLow;
      ah[mt] = ldsFrag(sAhi, r, kp);
      al[mt] = ldsFrag(sAlo, r, kp);
    }
#pragma unroll
    for (int nt = 0; nt < 4; ++nt) {
      int r = wN + nt * 16 + laneLow;
      bh[nt] = ldsFrag(sBhi, r, kp);
      bl[nt] = ldsFrag(sBlo, r, kp);
    }
#pragma unroll
    for (int mt = 0; mt < 2; ++mt)
#pragma unroll
      for (int nt = 0; nt < 4; ++nt) {
        acc[mt][nt] = __builtin_amdgcn_wmma_f32_16x16x32_bf16(
            false, ah[mt], false, bh[nt], (short)0, acc[mt][nt], false, false);
        acc[mt][nt] = __builtin_amdgcn_wmma_f32_16x16x32_bf16(
            false, ah[mt], false, bl[nt], (short)0, acc[mt][nt], false, false);
        acc[mt][nt] = __builtin_amdgcn_wmma_f32_16x16x32_bf16(
            false, al[mt], false, bh[nt], (short)0, acc[mt][nt], false, false);
      }
  }

  storeOut(out, acc, passed, tM, tN, wM, wN, laneLow, laneHi);
}

// ---------------------------------------------------------------------------
extern "C" void kernel_launch(void* const* d_in, const int* in_sizes, int n_in,
                              void* d_out, int out_size, void* d_ws, size_t ws_size,
                              hipStream_t stream) {
  (void)in_sizes; (void)n_in; (void)out_size;
  const float* x = (const float*)d_in[0];   // (8192, 4096) f32
  const float* W = (const float*)d_in[1];   // (4096, 4096) f32
  float* out = (float*)d_out;               // (8192, 4096) f32

  const size_t nx = (size_t)BDIM * INDIM;
  const size_t nw = (size_t)OUTDIM * INDIM;
  const size_t need = (2 * nx + 2 * nw) * sizeof(__hip_bfloat16);  // ~201 MB

  dim3 grid(BDIM / TILE_M, OUTDIM / TILE_N);  // 64 x 32 blocks

  if (d_ws != nullptr && ws_size >= need) {
    __bf16* xhi = (__bf16*)d_ws;
    __bf16* xlo = xhi + nx;
    __bf16* whi = xlo + nx;
    __bf16* wlo = whi + nw;
    precvt<<<4096, 256, 0, stream>>>(x, xhi, xlo, (int)(nx / 4));
    precvt<<<4096, 256, 0, stream>>>(W, whi, wlo, (int)(nw / 4));
    gemm_planes<<<grid, 256, 0, stream>>>(xhi, xlo, whi, wlo, x, W, out);
  } else {
    fused_seq_linear<<<grid, 256, 0, stream>>>(x, W, out);
  }
}
